// SequentialReachingNetwork_3753801417164
// MI455X (gfx1250) — matrix operand order repaired and checked by reading, not compiled, not run on hardware
//
#include <hip/hip_runtime.h>
#include <hip/hip_bf16.h>

typedef _Float16 v8h  __attribute__((ext_vector_type(8)));
typedef _Float16 v16h __attribute__((ext_vector_type(16)));
typedef float    v8f  __attribute__((ext_vector_type(8)));

#define NB 2048   // batch (GEMM M)
#define NH 1024   // hidden (GEMM N and K)
#define NT 512    // duration

// ---------------------------------------------------------------------------
// Fast branch-free tanh: native v_tanh_f32 if the builtin exists, else
// tanh(x) = sign(x) * (1 - e^{-2|x|}) / (1 + e^{-2|x|}) via v_exp_f32 + v_rcp_f32
// ---------------------------------------------------------------------------
__device__ __forceinline__ float fast_tanh(float x) {
#if __has_builtin(__builtin_amdgcn_tanhf)
  return __builtin_amdgcn_tanhf(x);
#else
  float ax = __builtin_fabsf(x);
  float e  = __builtin_amdgcn_exp2f(ax * -2.88539008177792681472f); // e^{-2ax}
  float r  = (1.f - e) * __builtin_amdgcn_rcpf(1.f + e);
  return __builtin_copysignf(r, x);
#endif
}

// ---------------------------------------------------------------------------
// CDNA5 async global -> LDS copy of 32 contiguous bytes (2 x b128), ASYNCcnt
// ---------------------------------------------------------------------------
__device__ __forceinline__ void async_copy32(const _Float16* g, _Float16* s) {
  asm volatile(
      "global_load_async_to_lds_b128 %0, %1, off\n\t"
      "global_load_async_to_lds_b128 %0, %1, off offset:16"
      :
      : "v"((uint32_t)(uintptr_t)s), "v"((uint64_t)(uintptr_t)g)
      : "memory");
}
__device__ __forceinline__ void wait_async0() {
  asm volatile("s_wait_asynccnt 0x0" ::: "memory");
}

// ---------------------------------------------------------------------------
// One-time prologue: tiny MLP 8->25->50->10 per batch row
// ---------------------------------------------------------------------------
__global__ __launch_bounds__(256) void mlp_kernel(
    const float* __restrict__ tgt,
    const float* __restrict__ W1, const float* __restrict__ b1,
    const float* __restrict__ W2, const float* __restrict__ b2,
    const float* __restrict__ W3, const float* __restrict__ b3,
    float* __restrict__ rnn_in)
{
  int b = blockIdx.x * blockDim.x + threadIdx.x;
  if (b >= NB) return;
  float in[8];
  #pragma unroll
  for (int j = 0; j < 6; ++j) in[j] = tgt[b * 6 + j];
  in[6] = 0.f; in[7] = 0.f;

  float h1[25];
  #pragma unroll
  for (int o = 0; o < 25; ++o) {
    float s = b1[o];
    #pragma unroll
    for (int i = 0; i < 8; ++i) s += in[i] * W1[i * 25 + o];
    h1[o] = s > 0.f ? s : 0.f;
  }
  float h2[50];
  for (int o = 0; o < 50; ++o) {
    float s = b2[o];
    #pragma unroll
    for (int i = 0; i < 25; ++i) s += h1[i] * W2[i * 50 + o];
    h2[o] = s > 0.f ? s : 0.f;
  }
  #pragma unroll
  for (int o = 0; o < 10; ++o) {
    float s = b3[o];
    #pragma unroll
    for (int i = 0; i < 50; ++i) s += h2[i] * W3[i * 10 + o];
    rnn_in[b * 10 + o] = s;
  }
}

// ---------------------------------------------------------------------------
// One-time: u = rnn_in @ W_ih + b_h  (fp32), and h_1 = tanh(u) (f16, h0 = 0)
// grid (NH/256, NB)
// ---------------------------------------------------------------------------
__global__ __launch_bounds__(256) void drive_kernel(
    const float* __restrict__ rnn_in, const float* __restrict__ W_ih,
    const float* __restrict__ b_h,
    float* __restrict__ u, _Float16* __restrict__ h1out)
{
  int col = blockIdx.x * 256 + threadIdx.x;
  int b   = blockIdx.y;
  float s = b_h[col];
  #pragma unroll
  for (int i = 0; i < 10; ++i) s += rnn_in[b * 10 + i] * W_ih[i * NH + col];
  u[(size_t)b * NH + col]     = s;
  h1out[(size_t)b * NH + col] = (_Float16)fast_tanh(s);
}

// ---------------------------------------------------------------------------
// One-time: W_hh fp32 [K][N] -> f16 transposed wT [N][K]
// (so B-tile staging in the step kernel is row-contiguous -> pure async b128)
// ---------------------------------------------------------------------------
__global__ __launch_bounds__(256) void wconv_kernel(
    const float* __restrict__ w, _Float16* __restrict__ wT)
{
  int idx = blockIdx.x * 256 + threadIdx.x;   // over NH*NH
  int k = idx >> 10, n = idx & (NH - 1);
  wT[(size_t)n * NH + k] = (_Float16)w[idx];
}

// ---------------------------------------------------------------------------
// Readout of one batch row: pos = h[row,:] @ Wout  (Wout: [NH][2] fp32)
// ---------------------------------------------------------------------------
__device__ inline void readout_row(const _Float16* __restrict__ h,
                                   const float* __restrict__ wout,
                                   int row, float* __restrict__ pos)
{
  const v8h* hp = reinterpret_cast<const v8h*>(h + (size_t)row * NH);
  float px = 0.f, py = 0.f;
  for (int kk = 0; kk < NH / 8; ++kk) {
    v8h hv = hp[kk];
    const float4* wp = reinterpret_cast<const float4*>(wout + kk * 16);
    #pragma unroll
    for (int j = 0; j < 4; ++j) {
      float4 q = wp[j];               // {w[2j][0], w[2j][1], w[2j+1][0], w[2j+1][1]}
      px += (float)hv[2 * j] * q.x + (float)hv[2 * j + 1] * q.z;
      py += (float)hv[2 * j] * q.y + (float)hv[2 * j + 1] * q.w;
    }
  }
  pos[row * 2 + 0] = px;
  pos[row * 2 + 1] = py;
}

// ---------------------------------------------------------------------------
// One RNN step:  h_out = tanh(h_in @ W + u)   via v_wmma_f32_16x16x32_f16
// Block tile 128x128, 8 waves (4Mx2N), wave tile 32x64 = 2x4 WMMA tiles.
// Double-buffered LDS, tiles staged with global_load_async_to_lds_b128,
// one s_wait_asynccnt + barrier per K-iteration.
// Blocks with blockIdx.y==0 also emit pos for the *input* state h_in.
// ---------------------------------------------------------------------------
__global__ __launch_bounds__(256) void step_kernel(
    const _Float16* __restrict__ h_in,   // [NB][NH] f16
    const _Float16* __restrict__ wT,     // [NH][NH] f16, TRANSPOSED: wT[n][k]
    const float*    __restrict__ u,      // [NB][NH] fp32
    const float*    __restrict__ wout,   // [NH][2]  fp32
    _Float16*       __restrict__ h_out,  // [NB][NH] f16
    float*          __restrict__ pos_out)// [NB][2]  fp32 (prev scan index)
{
  __shared__ _Float16 As[2][128][32];  // A tiles (row-major)      2 x 8 KB
  __shared__ _Float16 Bt[2][128][32];  // B tiles ([n][k])         2 x 8 KB

  const int tid  = threadIdx.x;
  const int lane = tid & 31;
  const int wave = tid >> 5;
  const int ll   = lane & 15;
  const int lh   = lane >> 4;
  const int m0   = blockIdx.x * 128;
  const int n0   = blockIdx.y * 128;
  const int wr0  = (wave >> 1) * 32;   // wave row origin (4 M-strips)
  const int wc0  = (wave & 1) * 64;    // wave col origin (2 N-strips)

  // staging decomposition: each thread owns one 16-half (32 B) row chunk
  const int sr = tid >> 1, sc = (tid & 1) * 16;

  v8f acc[2][4] = {};

  // stage tile kb into LDS buffer (kb & 1)
  auto stage = [&](int kb) {
    const int k = kb * 32, bf = kb & 1;
    async_copy32(h_in + (size_t)(m0 + sr) * NH + k + sc, &As[bf][sr][sc]);
    async_copy32(wT   + (size_t)(n0 + sr) * NH + k + sc, &Bt[bf][sr][sc]);
  };

  stage(0);
  wait_async0();
  __syncthreads();

  for (int kb = 0; kb < NH / 32; ++kb) {
    const int bf = kb & 1;
    if (kb + 1 < NH / 32) stage(kb + 1);   // DMA next tiles during compute

    // Build fragments per ISA 7.12.2 16-bit layouts
    v16h afrag[2], bfrag[4];
    #pragma unroll
    for (int ti = 0; ti < 2; ++ti) {
      const _Float16* ap = &As[bf][wr0 + ti * 16 + ll][lh * 8];
      v8h a0 = *reinterpret_cast<const v8h*>(ap);        // K = lh*8 .. +7
      v8h a1 = *reinterpret_cast<const v8h*>(ap + 16);   // K = lh*8+16 .. +23
      #pragma unroll
      for (int j = 0; j < 8; ++j) { afrag[ti][j] = a0[j]; afrag[ti][j + 8] = a1[j]; }
    }
    #pragma unroll
    for (int tj = 0; tj < 4; ++tj) {
      const _Float16* bp = &Bt[bf][wc0 + tj * 16 + ll][lh * 8]; // K = lh*8..+7, lh*8+16..+23
      v8h b0 = *reinterpret_cast<const v8h*>(bp);
      v8h b1 = *reinterpret_cast<const v8h*>(bp + 16);
      #pragma unroll
      for (int j = 0; j < 8; ++j) { bfrag[tj][j] = b0[j]; bfrag[tj][j + 8] = b1[j]; }
    }
    #pragma unroll
    for (int ti = 0; ti < 2; ++ti)
      #pragma unroll
      for (int tj = 0; tj < 4; ++tj)
        acc[ti][tj] = __builtin_amdgcn_wmma_f32_16x16x32_f16(
            false, afrag[ti], false, bfrag[tj],
            (short)0, acc[ti][tj], false, false);

    wait_async0();      // next tiles landed (had the whole WMMA burst to fly)
    __syncthreads();
  }

  // epilogue: + drive, tanh, f16 store  (C layout: row = r + lh*8, col = ll)
  #pragma unroll
  for (int ti = 0; ti < 2; ++ti) {
    #pragma unroll
    for (int tj = 0; tj < 4; ++tj) {
      const int gc = n0 + wc0 + tj * 16 + ll;
      #pragma unroll
      for (int r = 0; r < 8; ++r) {
        const int gr = m0 + wr0 + ti * 16 + lh * 8 + r;
        float v = acc[ti][tj][r] + u[(size_t)gr * NH + gc];
        h_out[(size_t)gr * NH + gc] = (_Float16)fast_tanh(v);
      }
    }
  }

  // fused readout of the INPUT state (pos for the previous scan index)
  if (blockIdx.y == 0 && tid < 128)
    readout_row(h_in, wout, m0 + tid, pos_out);
}

// ---------------------------------------------------------------------------
// Tail: pos[511] from the final state
// ---------------------------------------------------------------------------
__global__ __launch_bounds__(256) void readout_kernel(
    const _Float16* __restrict__ h, const float* __restrict__ wout,
    float* __restrict__ pos)
{
  int row = blockIdx.x * 256 + threadIdx.x;
  readout_row(h, wout, row, pos);
}

// ---------------------------------------------------------------------------
extern "C" void kernel_launch(void* const* d_in, const int* in_sizes, int n_in,
                              void* d_out, int out_size, void* d_ws, size_t ws_size,
                              hipStream_t stream)
{
  const float* tgt  = (const float*)d_in[0];
  const float* W1   = (const float*)d_in[1];
  const float* b1   = (const float*)d_in[2];
  const float* W2   = (const float*)d_in[3];
  const float* b2   = (const float*)d_in[4];
  const float* W3   = (const float*)d_in[5];
  const float* b3   = (const float*)d_in[6];
  const float* W_ih = (const float*)d_in[7];
  const float* W_hh = (const float*)d_in[8];
  const float* b_h  = (const float*)d_in[9];
  const float* Wout = (const float*)d_in[10];
  float* pos = (float*)d_out;               // [512][2048][2] fp32

  // workspace layout
  char* ws = (char*)d_ws;
  float*    u   = (float*)(ws);                       //  8 MB fp32 drive
  _Float16* hb0 = (_Float16*)(ws + 8388608);          //  4 MB f16 state ping
  _Float16* hb1 = (_Float16*)(ws + 12582912);         //  4 MB f16 state pong
  _Float16* wT  = (_Float16*)(ws + 16777216);         //  2 MB f16 W_hh^T
  float*    rnn = (float*)(ws + 18874368);            // 80 KB rnn_in

  // prologue (once)
  mlp_kernel<<<NB / 256, 256, 0, stream>>>(tgt, W1, b1, W2, b2, W3, b3, rnn);
  wconv_kernel<<<(NH * NH) / 256, 256, 0, stream>>>(W_hh, wT);
  drive_kernel<<<dim3(NH / 256, NB), 256, 0, stream>>>(rnn, W_ih, b_h, u, hb1);
  // hb1 now holds h_1 = tanh(u)   (since h_0 = 0)

  // steps i = 1..511: h_{i+1} = tanh(h_i @ W + u); also pos[i-1] = h_i @ Wout
  for (int i = 1; i < NT; ++i) {
    const _Float16* hin = (i & 1) ? hb1 : hb0;
    _Float16*      hout = (i & 1) ? hb0 : hb1;
    step_kernel<<<dim3(NB / 128, NH / 128), 256, 0, stream>>>(
        hin, wT, u, Wout, hout, pos + (size_t)(i - 1) * NB * 2);
  }
  // pos[511] = h_512 @ Wout   (h_512 lives in hb0 after step 511)
  readout_kernel<<<NB / 256, 256, 0, stream>>>(hb0, Wout,
                                               pos + (size_t)(NT - 1) * NB * 2);
}
